// ReversibleLSTM_9002251453153
// MI455X (gfx1250) — matrix operand order repaired and checked by reading, not compiled, not run on hardware
//
#include <hip/hip_runtime.h>

// ---------------------------------------------------------------------------
// Reversed LSTM on MI455X (gfx1250, wave32, WMMA).  B=32, T=4096, I=H=128.
//
// Phase 0 (pack):  W_ih/W_hh -> f16 WMMA B-fragment order; bias = b_ih+b_hh.
// Phase 1 (xproj): time-parallel xg[t] = x_t @ W_ih^T + bias via WMMA,
//                  stored f16 in C-fragment slot order [t][blk][wv][lane][j][r]
//                  so the scan loads accumulator init with 4 coalesced b128s.
// Phase 2 (scan):  latency-bound 4096-step recurrence, 2 blocks x 16 batches,
//                  8 waves; wave w owns N-tiles {w,w+8,w+16,w+24} so acc0..3
//                  hold i/f/g/o for the same (batch,col) at the same slot ->
//                  gate math is pure register VALU.  W_hh fragments fully
//                  resident (128 VGPRs, no weight traffic in loop), h tile
//                  double-buffered in LDS, ONE barrier per step, xg for t-1
//                  register-prefetched during step t.  Total pressure < 256.
// ---------------------------------------------------------------------------

typedef __attribute__((ext_vector_type(16))) _Float16     v16h;
typedef __attribute__((ext_vector_type(8)))  float        v8f;
typedef __attribute__((ext_vector_type(8)))  unsigned int v8u;

#define B_   32
#define T_   4096
#define I_   128
#define H_   128
#define G_   512

#define HSTR 136   // f16 LDS row stride (halves), padded vs 128

// ws layout: [0,256KB) wpack dwords | [256KB,258KB) bias | [1MB, 1MB+128MB) xg f16
#define NPACK 65536
#define NBIAS 512
#define XG_BYTE_OFF (1u << 20)

__device__ __forceinline__ unsigned int pack2h(float lo, float hi) {
  _Float16 l = (_Float16)lo, h = (_Float16)hi;
  return ((unsigned int)__builtin_bit_cast(unsigned short, h) << 16) |
         (unsigned int)__builtin_bit_cast(unsigned short, l);
}
__device__ __forceinline__ float lo_h(unsigned int dw) {
  return (float)__builtin_bit_cast(_Float16, (unsigned short)(dw & 0xffffu));
}
__device__ __forceinline__ float hi_h(unsigned int dw) {
  return (float)__builtin_bit_cast(_Float16, (unsigned short)(dw >> 16));
}

__global__ void rlstm_pack_kernel(const float* __restrict__ Wih,
                                  const float* __restrict__ Whh,
                                  const float* __restrict__ bih,
                                  const float* __restrict__ bhh,
                                  unsigned int* __restrict__ wpack,
                                  float* __restrict__ bias) {
  int tid = blockIdx.x * blockDim.x + threadIdx.x;
  if (tid < NPACK) {
    const float* W = (tid < 32768) ? Wih : Whh;
    int d      = tid & 32767;
    int fragid = d >> 8;            // ntile*4 + kstep, 0..127
    int lane   = (d >> 3) & 31;
    int v      = d & 7;
    int ntile  = fragid >> 2;
    int kstep  = fragid & 3;
    // 16-bit B-matrix (32x16) layout: lanes 0-15: K=0..7 (v0-3), 16..23 (v4-7); lanes 16-31: +8
    int n = ntile * 16 + (lane & 15);
    int k = kstep * 32 + ((lane >> 4) << 3) + ((v >> 2) << 4) + ((v & 3) << 1);
    wpack[tid] = pack2h(W[n * I_ + k], W[n * I_ + k + 1]);   // B[k][n] = W[n][k]
  } else if (tid < NPACK + NBIAS) {
    int g = tid - NPACK;
    bias[g] = bih[g] + bhh[g];
  }
}

__device__ __forceinline__ float fast_sigmoid(float v) {
  return 1.0f / (1.0f + __expf(-v));
}
__device__ __forceinline__ float fast_tanh(float v) {
  return 2.0f / (1.0f + __expf(-2.0f * v)) - 1.0f;
}

// 16-bit A fragment (16x32 layout) from a padded f16 LDS tile.
// lanes 0-15: K = 0..7 (v0-3), 16..23 (v4-7); lanes 16-31: +8 / +24.
__device__ __forceinline__ v16h a_frag(const _Float16* tile, int am, int khalf,
                                       int ks) {
  int base = am * HSTR + khalf + ks * 32;
  uint4 lo = *(const uint4*)(tile + base);
  uint4 hi = *(const uint4*)(tile + base + 16);
  v8u au;
  au[0] = lo.x; au[1] = lo.y; au[2] = lo.z; au[3] = lo.w;
  au[4] = hi.x; au[5] = hi.y; au[6] = hi.z; au[7] = hi.w;
  return __builtin_bit_cast(v16h, au);
}

// ---------------------------------------------------------------------------
// Phase 1: xg[t][blk] = x[blk*16..+15, t, :] @ W_ih^T + bias, f16, fragment
// slot order: dword index (((t*2+blk)*8 + wv)*32 + lane)*16 + j*4 + r/2.
// ---------------------------------------------------------------------------
__global__ __launch_bounds__(256, 1)
void rlstm_xproj_kernel(const float* __restrict__ x,
                        const unsigned int* __restrict__ wpack,
                        const float* __restrict__ bias,
                        unsigned int* __restrict__ xg) {
  __shared__ _Float16 xh[16 * HSTR];

  const int t    = blockIdx.x >> 1;
  const int blk  = blockIdx.x & 1;
  const int tid  = threadIdx.x;
  const int lane = tid & 31;
  const int wv   = tid >> 5;
  const int bb   = blk * 16;

  const int em = tid >> 4;
  const int ek = (tid & 15) << 3;
  const int am    = lane & 15;
  const int khalf = (lane >> 4) * 8;
  const int hcol  = wv * 16 + am;

  // stage x tile (16 batches x 128) -> f16 LDS
  {
    const float* xr = x + ((size_t)(bb + em) * T_ + t) * I_ + ek;
    float4 a0 = *(const float4*)xr;
    float4 a1 = *(const float4*)(xr + 4);
    uint4 pk;
    pk.x = pack2h(a0.x, a0.y); pk.y = pack2h(a0.z, a0.w);
    pk.z = pack2h(a1.x, a1.y); pk.w = pack2h(a1.z, a1.w);
    *(uint4*)(&xh[em * HSTR + ek]) = pk;
  }
  __syncthreads();

  v8f acc[4];
#pragma unroll
  for (int j = 0; j < 4; ++j)
#pragma unroll
    for (int r = 0; r < 8; ++r) acc[j][r] = 0.0f;

#pragma unroll
  for (int ks = 0; ks < 4; ++ks) {
    v16h a = a_frag(xh, am, khalf, ks);
#pragma unroll
    for (int j = 0; j < 4; ++j) {
      int nt = wv + 8 * j;
      v16h b = __builtin_bit_cast(
          v16h, *(const v8u*)(wpack + (nt * 4 + ks) * 256 + lane * 8));
      acc[j] = __builtin_amdgcn_wmma_f32_16x16x32_f16(
          false, a, false, b, (short)0, acc[j], false, false);
    }
  }

  // + bias, pack to f16, store in scan order (64B contiguous per lane)
  unsigned int* dst =
      xg + ((size_t)(((t * 2 + blk) * 8 + wv) * 32 + lane)) * 16;
#pragma unroll
  for (int j = 0; j < 4; ++j) {
    float bv = bias[j * 128 + hcol];
    uint4 q;
    q.x = pack2h(acc[j][0] + bv, acc[j][1] + bv);
    q.y = pack2h(acc[j][2] + bv, acc[j][3] + bv);
    q.z = pack2h(acc[j][4] + bv, acc[j][5] + bv);
    q.w = pack2h(acc[j][6] + bv, acc[j][7] + bv);
    *(uint4*)(dst + j * 4) = q;
  }
}

// ---------------------------------------------------------------------------
// Phase 2: the sequential scan.
// ---------------------------------------------------------------------------
__global__ __launch_bounds__(256, 1)
void rlstm_scan_kernel(const unsigned int* __restrict__ xg,
                       const unsigned int* __restrict__ wpack,
                       float* __restrict__ out) {
  __shared__ _Float16 hb2[2][16 * HSTR];   // h tile, f16, double-buffered

  const int tid  = threadIdx.x;
  const int lane = tid & 31;
  const int wv   = tid >> 5;             // wave 0..7
  const int blk  = blockIdx.x;
  const int bb   = blk * 16;

  const int am    = lane & 15;           // A-fragment row (M)
  const int khalf = (lane >> 4) * 8;     // A-fragment K sub-offset
  const int hcol  = wv * 16 + am;        // owned hidden column
  const int mhi   = (lane >> 4) * 8;     // C-fragment row offset

  // ---- resident W_hh fragments: ntile = wv + 8j (j = gate segment) ----
  v16h bHH[4][4];
#pragma unroll
  for (int j = 0; j < 4; ++j) {
    int nt = wv + 8 * j;
#pragma unroll
    for (int ks = 0; ks < 4; ++ks)
      bHH[j][ks] = __builtin_bit_cast(
          v16h, *(const v8u*)(wpack + (128 + nt * 4 + ks) * 256 + lane * 8));
  }

  // ---- init: h = 0 in both LDS buffers, c = 0 ----
  for (int i = tid; i < (2 * 16 * HSTR) / 2; i += 256)
    ((unsigned int*)hb2)[i] = 0u;
  float c0[8];
#pragma unroll
  for (int r = 0; r < 8; ++r) c0[r] = 0.0f;

  // per-lane xg base (dwords); step stride = 2*8*32*16 dwords
  const unsigned int* xgbase = xg + ((size_t)(blk * 8 + wv) * 32 + lane) * 16;
  const size_t xstride = (size_t)2 * 8 * 32 * 16;

  // prefetch xg for t = T-1
  uint4 xq[4];
  {
    const unsigned int* p = xgbase + (size_t)(T_ - 1) * xstride;
#pragma unroll
    for (int j = 0; j < 4; ++j) xq[j] = *(const uint4*)(p + j * 4);
  }

  __syncthreads();                        // h zeros visible

  for (int t = T_ - 1; t >= 0; --t) {
    const int p = t & 1;
    const _Float16* hcur = hb2[p];
    _Float16*       hnxt = hb2[p ^ 1];

    // ---- acc init from prefetched xg (x@Wih^T + bias, f16 -> f32) ----
    v8f acc[4];
#pragma unroll
    for (int j = 0; j < 4; ++j) {
      acc[j][0] = lo_h(xq[j].x); acc[j][1] = hi_h(xq[j].x);
      acc[j][2] = lo_h(xq[j].y); acc[j][3] = hi_h(xq[j].y);
      acc[j][4] = lo_h(xq[j].z); acc[j][5] = hi_h(xq[j].z);
      acc[j][6] = lo_h(xq[j].w); acc[j][7] = hi_h(xq[j].w);
    }

    // ---- prefetch xg for t-1 (consumed next iteration) ----
    if (t > 0) {
      const unsigned int* pp = xgbase + (size_t)(t - 1) * xstride;
#pragma unroll
      for (int j = 0; j < 4; ++j) xq[j] = *(const uint4*)(pp + j * 4);
    }

    // ---- h-GEMM on fully resident W_hh ----
#pragma unroll
    for (int ks = 0; ks < 4; ++ks) {
      v16h a = a_frag(hcur, am, khalf, ks);
#pragma unroll
      for (int j = 0; j < 4; ++j)
        acc[j] = __builtin_amdgcn_wmma_f32_16x16x32_f16(
            false, a, false, bHH[j][ks], (short)0, acc[j], false, false);
    }

    // ---- gate math directly on accumulators (i,f,g,o = acc0..acc3) ----
#pragma unroll
    for (int r = 0; r < 8; ++r) {
      float ig = fast_sigmoid(acc[0][r]);
      float fg = fast_sigmoid(acc[1][r]);
      float gt = fast_tanh   (acc[2][r]);
      float og = fast_sigmoid(acc[3][r]);
      c0[r] = fg * c0[r] + ig * gt;
      float hv = og * fast_tanh(c0[r]);
      int m = mhi + r;
      out[((size_t)(bb + m) * T_ + t) * H_ + hcol] = hv;   // y[b,t,hcol]
      hnxt[m * HSTR + hcol] = (_Float16)hv;                // h for step t-1
    }

    __syncthreads();                      // single barrier per step
  }
}

extern "C" void kernel_launch(void* const* d_in, const int* in_sizes, int n_in,
                              void* d_out, int out_size, void* d_ws, size_t ws_size,
                              hipStream_t stream) {
  const float* x   = (const float*)d_in[0];
  const float* Wih = (const float*)d_in[1];
  const float* Whh = (const float*)d_in[2];
  const float* bih = (const float*)d_in[3];
  const float* bhh = (const float*)d_in[4];

  unsigned int* wpack = (unsigned int*)d_ws;
  float*        bias  = (float*)((char*)d_ws + (size_t)NPACK * 4);
  unsigned int* xgbuf = (unsigned int*)((char*)d_ws + XG_BYTE_OFF);

  int prep_threads = NPACK + NBIAS;                 // 66048
  rlstm_pack_kernel<<<(prep_threads + 255) / 256, 256, 0, stream>>>(
      Wih, Whh, bih, bhh, wpack, bias);

  rlstm_xproj_kernel<<<T_ * 2, 256, 0, stream>>>(x, wpack, bias, xgbuf);

  rlstm_scan_kernel<<<2, 256, 0, stream>>>(xgbuf, wpack, (float*)d_out);
}
